// STSN_33371895890635
// MI455X (gfx1250) — compile-verified, matrix operand order checked by reading, not run on hardware
//
#include <hip/hip_runtime.h>
#include <hip/hip_bf16.h>

typedef _Float16 f16;
typedef _Float16 v16h __attribute__((ext_vector_type(16)));
typedef _Float16 v8h  __attribute__((ext_vector_type(8)));
typedef float    v8f  __attribute__((ext_vector_type(8)));
typedef int      v4i  __attribute__((vector_size(16)));   // matches builtin 'V4i'

// Probe for CDNA5 async global->LDS path; fall back to sync loads if absent.
#if defined(__has_builtin)
#if __has_builtin(__builtin_amdgcn_global_load_async_to_lds_b128) && \
    __has_builtin(__builtin_amdgcn_s_wait_asynccnt)
#define STSN_USE_ASYNC 1
#endif
#endif
#ifndef STSN_USE_ASYNC
#define STSN_USE_ASYNC 0
#endif

#define STSN_GPTR(p) ((__attribute__((address_space(1))) v4i*)(p))
#define STSN_LPTR(p) ((__attribute__((address_space(3))) v4i*)(p))

// Problem constants (from reference)
static constexpr int Hc = 48, Wc = 48, Bc = 2;
static constexpr int HWc = Hc * Wc;          // 2304
static constexpr int NT = Bc * HWc;          // 4608  (GEMM N, multiple of 128)

// ---------------------------------------------------------------------------
// WMMA GEMM:  C[M][N] = sum_k A[M][K] * B[N][K]   (A,B f16 K-contiguous, C f32)
// M % 128 == 0, N % 128 == 0, K % 32 == 0 for all calls in this pipeline.
// Block: 256 threads = 8 waves, tile 128x128, BK=32. Wave tile 64(M) x 32(N).
// Async path: double-buffered GLOBAL_LOAD_ASYNC_TO_LDS_B128 overlapped with WMMA.
// ---------------------------------------------------------------------------
static constexpr int BM = 128, BN = 128, BK = 32, LDP = BK + 8; // 80B pitch, 16B aligned

__global__ __launch_bounds__(256) void stsn_wmma_gemm(
    const f16* __restrict__ A, const f16* __restrict__ Bm, float* __restrict__ C,
    int M, int N, int K)
{
    const int tid  = threadIdx.x;
    const int wave = tid >> 5;
    const int lane = tid & 31;
    const int g    = lane >> 4;      // half-wave group
    const int r    = lane & 15;
    const int mblk = blockIdx.y * BM;
    const int nblk = blockIdx.x * BN;
    const int wm0  = (wave >> 2) * 64;   // 0 / 64
    const int wn0  = (wave & 3) * 32;    // 0..96

    v8f acc[4][2];
#pragma unroll
    for (int i = 0; i < 4; ++i)
#pragma unroll
        for (int j = 0; j < 2; ++j)
#pragma unroll
            for (int e = 0; e < 8; ++e) acc[i][j][e] = 0.f;

#if STSN_USE_ASYNC
    __shared__ f16 As[2][BM][LDP];
    __shared__ f16 Bs[2][BN][LDP];

    auto issue_tile = [&](int k0, int buf) {
#pragma unroll
        for (int i = 0; i < 2; ++i) {
            int idx = tid + i * 256;           // 0..511
            int row = idx >> 2;
            int seg = idx & 3;
            __builtin_amdgcn_global_load_async_to_lds_b128(
                STSN_GPTR(A + (size_t)(mblk + row) * K + k0 + seg * 8),
                STSN_LPTR(&As[buf][row][seg * 8]), 0, 0);
            __builtin_amdgcn_global_load_async_to_lds_b128(
                STSN_GPTR(Bm + (size_t)(nblk + row) * K + k0 + seg * 8),
                STSN_LPTR(&Bs[buf][row][seg * 8]), 0, 0);
        }
    };

    issue_tile(0, 0);
    __builtin_amdgcn_s_wait_asynccnt(0);
    __syncthreads();

    const int nk = K / BK;
    for (int it = 0; it < nk; ++it) {
        const int cur = it & 1;
        if (it + 1 < nk) issue_tile((it + 1) * BK, cur ^ 1);

        v16h af[4], bf[2];
#pragma unroll
        for (int wm = 0; wm < 4; ++wm) {
            int row = wm0 + wm * 16 + r;
            v8h lo = *(const v8h*)&As[cur][row][g * 8];
            v8h hi = *(const v8h*)&As[cur][row][16 + g * 8];
            af[wm] = __builtin_shufflevector(lo, hi, 0,1,2,3,4,5,6,7,8,9,10,11,12,13,14,15);
        }
#pragma unroll
        for (int wn = 0; wn < 2; ++wn) {
            int row = wn0 + wn * 16 + r;
            v8h lo = *(const v8h*)&Bs[cur][row][g * 8];
            v8h hi = *(const v8h*)&Bs[cur][row][16 + g * 8];
            bf[wn] = __builtin_shufflevector(lo, hi, 0,1,2,3,4,5,6,7,8,9,10,11,12,13,14,15);
        }
#pragma unroll
        for (int wm = 0; wm < 4; ++wm)
#pragma unroll
            for (int wn = 0; wn < 2; ++wn)
                acc[wm][wn] = __builtin_amdgcn_wmma_f32_16x16x32_f16(
                    false, af[wm], false, bf[wn], (short)0, acc[wm][wn], false, false);

        __builtin_amdgcn_s_wait_asynccnt(0);
        __syncthreads();
    }
#else
    __shared__ f16 As[BM][LDP];
    __shared__ f16 Bs[BN][LDP];
    for (int k0 = 0; k0 < K; k0 += BK) {
#pragma unroll
        for (int i = 0; i < 2; ++i) {
            int idx = tid + i * 256;           // 0..511
            int row = idx >> 2;
            int seg = idx & 3;
            const float4 av = *(const float4*)(A  + (size_t)(mblk + row) * K + k0 + seg * 8);
            const float4 bv = *(const float4*)(Bm + (size_t)(nblk + row) * K + k0 + seg * 8);
            *(float4*)&As[row][seg * 8] = av;
            *(float4*)&Bs[row][seg * 8] = bv;
        }
        __syncthreads();

        v16h af[4], bf[2];
#pragma unroll
        for (int wm = 0; wm < 4; ++wm) {
            int row = wm0 + wm * 16 + r;
            v8h lo = *(const v8h*)&As[row][g * 8];
            v8h hi = *(const v8h*)&As[row][16 + g * 8];
            af[wm] = __builtin_shufflevector(lo, hi, 0,1,2,3,4,5,6,7,8,9,10,11,12,13,14,15);
        }
#pragma unroll
        for (int wn = 0; wn < 2; ++wn) {
            int row = wn0 + wn * 16 + r;
            v8h lo = *(const v8h*)&Bs[row][g * 8];
            v8h hi = *(const v8h*)&Bs[row][16 + g * 8];
            bf[wn] = __builtin_shufflevector(lo, hi, 0,1,2,3,4,5,6,7,8,9,10,11,12,13,14,15);
        }
#pragma unroll
        for (int wm = 0; wm < 4; ++wm)
#pragma unroll
            for (int wn = 0; wn < 2; ++wn)
                acc[wm][wn] = __builtin_amdgcn_wmma_f32_16x16x32_f16(
                    false, af[wm], false, bf[wn], (short)0, acc[wm][wn], false, false);
        __syncthreads();
    }
#endif

    // C/D layout: lane r = column, VGPR e = row e (+8 for upper half-wave)
#pragma unroll
    for (int wm = 0; wm < 4; ++wm)
#pragma unroll
        for (int wn = 0; wn < 2; ++wn) {
            int n = nblk + wn0 + wn * 16 + r;
            int mbase = mblk + wm0 + wm * 16 + g * 8;
#pragma unroll
            for (int e = 0; e < 8; ++e)
                C[(size_t)(mbase + e) * N + n] = acc[wm][wn][e];
        }
}

// ---------------------------------------------------------------------------
// Build concat feature f[1024][B][HW]: ch<512 = support, ch>=512 = reference.
// swap=0: support=reference (tt pass). swap=1: support = roll(datas,1) (tk).
// ---------------------------------------------------------------------------
__global__ void stsn_build_concat(const float* __restrict__ datas, float* __restrict__ f, int swap)
{
    int idx = blockIdx.x * blockDim.x + threadIdx.x;
    if (idx >= 1024 * NT) return;
    int hw = idx % HWc;
    int t  = idx / HWc;
    int b  = t % Bc;
    int cf = t / Bc;
    int c  = cf & 511;
    int sb = (cf < 512 && swap) ? (1 - b) : b;
    f[idx] = datas[((size_t)sb * 512 + c) * HWc + hw];
}

// ---------------------------------------------------------------------------
// Direct 3x3 conv, pad 1 (offset convs, M=18 — negligible FLOPs)
// X: [Cin][B][HW], W: [Cout][Cin][9], Y: [Cout][B][HW]
// ---------------------------------------------------------------------------
__global__ void stsn_conv3x3_direct(const float* __restrict__ X, const float* __restrict__ W,
                                    const float* __restrict__ bias, float* __restrict__ Y,
                                    int Cin, int Cout)
{
    int idx = blockIdx.x * blockDim.x + threadIdx.x;
    if (idx >= Cout * NT) return;
    int x  = idx % Wc;
    int y  = (idx / Wc) % Hc;
    int b  = (idx / HWc) % Bc;
    int oc = idx / (HWc * Bc);
    float acc = bias ? bias[oc] : 0.f;
    for (int c = 0; c < Cin; ++c) {
        const float* xp = X + ((size_t)c * Bc + b) * HWc;
        const float* wp = W + ((size_t)oc * Cin + c) * 9;
#pragma unroll
        for (int kt = 0; kt < 9; ++kt) {
            int yy = y + kt / 3 - 1;
            int xx = x + kt % 3 - 1;
            if (yy >= 0 && yy < Hc && xx >= 0 && xx < Wc)
                acc += wp[kt] * xp[yy * Wc + xx];
        }
    }
    Y[((size_t)oc * Bc + b) * HWc + y * Wc + x] = acc;
}

// ---------------------------------------------------------------------------
// Bilinear-sampled im2col for deform conv.
// src [Cin][B][HW], offs [18][B][HW] (ch = kt*2 + {0:dy,1:dx}),
// Bmat [NT][K] f16 with k = c*9 + kt.
// ---------------------------------------------------------------------------
__global__ void stsn_bilinear_im2col(const float* __restrict__ src, const float* __restrict__ offs,
                                     f16* __restrict__ Bmat, int Cin)
{
    int idx = blockIdx.x * blockDim.x + threadIdx.x;
    if (idx >= 9 * NT) return;
    int x  = idx % Wc;
    int y  = (idx / Wc) % Hc;
    int kt = (idx / HWc) % 9;
    int b  = idx / (HWc * 9);
    int hw = y * Wc + x;
    int n  = b * HWc + hw;
    int K  = Cin * 9;

    float offy = offs[((size_t)(kt * 2 + 0) * Bc + b) * HWc + hw];
    float offx = offs[((size_t)(kt * 2 + 1) * Bc + b) * HWc + hw];
    float py = (float)y + (float)(kt / 3 - 1) + offy;
    float px = (float)x + (float)(kt % 3 - 1) + offx;
    float y0f = floorf(py), x0f = floorf(px);
    int y0 = (int)y0f, x0 = (int)x0f;
    int y1 = y0 + 1,  x1 = x0 + 1;
    float wy1 = py - y0f, wx1 = px - x0f;
    float w00 = (1.f - wy1) * (1.f - wx1);
    float w01 = (1.f - wy1) * wx1;
    float w10 = wy1 * (1.f - wx1);
    float w11 = wy1 * wx1;
    bool vy0 = (y0 >= 0) & (y0 < Hc), vy1 = (y1 >= 0) & (y1 < Hc);
    bool vx0 = (x0 >= 0) & (x0 < Wc), vx1 = (x1 >= 0) & (x1 < Wc);
    w00 = (vy0 && vx0) ? w00 : 0.f;
    w01 = (vy0 && vx1) ? w01 : 0.f;
    w10 = (vy1 && vx0) ? w10 : 0.f;
    w11 = (vy1 && vx1) ? w11 : 0.f;
    int cy0 = y0 < 0 ? 0 : (y0 > Hc - 1 ? Hc - 1 : y0);
    int cy1 = y1 < 0 ? 0 : (y1 > Hc - 1 ? Hc - 1 : y1);
    int cx0 = x0 < 0 ? 0 : (x0 > Wc - 1 ? Wc - 1 : x0);
    int cx1 = x1 < 0 ? 0 : (x1 > Wc - 1 ? Wc - 1 : x1);
    int i00 = cy0 * Wc + cx0, i01 = cy0 * Wc + cx1;
    int i10 = cy1 * Wc + cx0, i11 = cy1 * Wc + cx1;

    f16* bp = Bmat + (size_t)n * K + kt;
    for (int c = 0; c < Cin; ++c) {
        const float* sp = src + ((size_t)c * Bc + b) * HWc;
        float v = w00 * sp[i00] + w01 * sp[i01] + w10 * sp[i10] + w11 * sp[i11];
        bp[(size_t)c * 9] = (f16)v;
    }
}

// Regular (pad 1) im2col for neck 3x3 conv.
__global__ void stsn_im2col_regular(const float* __restrict__ src, f16* __restrict__ Bmat, int Cin)
{
    int idx = blockIdx.x * blockDim.x + threadIdx.x;
    if (idx >= 9 * NT) return;
    int x  = idx % Wc;
    int y  = (idx / Wc) % Hc;
    int kt = (idx / HWc) % 9;
    int b  = idx / (HWc * 9);
    int n  = b * HWc + y * Wc + x;
    int K  = Cin * 9;
    int yy = y + kt / 3 - 1;
    int xx = x + kt % 3 - 1;
    bool inb = (yy >= 0) & (yy < Hc) & (xx >= 0) & (xx < Wc);
    int si = inb ? (yy * Wc + xx) : 0;
    f16* bp = Bmat + (size_t)n * K + kt;
    for (int c = 0; c < Cin; ++c) {
        float v = inb ? src[((size_t)c * Bc + b) * HWc + si] : 0.f;
        bp[(size_t)c * 9] = (f16)v;
    }
}

// f32 -> f16 flat copy (weights)
__global__ void stsn_to_f16(const float* __restrict__ X, f16* __restrict__ Y, long n)
{
    long i = (long)blockIdx.x * blockDim.x + threadIdx.x;
    if (i < n) Y[i] = (f16)X[i];
}

// X [C][NT] f32 -> Bmat [NT][C] f16  (im2col for 1x1 convs)
__global__ void stsn_to_f16_T(const float* __restrict__ X, f16* __restrict__ Bmat, int C)
{
    long i = (long)blockIdx.x * blockDim.x + threadIdx.x;
    if (i >= (long)C * NT) return;
    int n = (int)(i / C);
    int c = (int)(i % C);
    Bmat[i] = (f16)X[(size_t)c * NT + n];
}

// ---------------------------------------------------------------------------
// GroupNorm (32 groups) + LeakyReLU(0.01).  X,Y: [C][B][HW]. One block per (b, g).
// ---------------------------------------------------------------------------
__global__ __launch_bounds__(256) void stsn_gn_leaky(
    const float* __restrict__ X, float* __restrict__ Y,
    const float* __restrict__ gamma, const float* __restrict__ beta,
    int C, int chpg)
{
    int G  = C / chpg;
    int b  = blockIdx.x / G;
    int gi = blockIdx.x % G;
    int c0 = gi * chpg;
    int count = chpg * HWc;
    int tid = threadIdx.x;

    float s = 0.f, s2 = 0.f;
    for (int i = tid; i < count; i += 256) {
        int c  = c0 + i / HWc;
        int hw = i % HWc;
        float v = X[((size_t)c * Bc + b) * HWc + hw];
        s += v; s2 += v * v;
    }
    __shared__ float sh1[256], sh2[256];
    sh1[tid] = s; sh2[tid] = s2;
    __syncthreads();
    for (int st = 128; st > 0; st >>= 1) {
        if (tid < st) { sh1[tid] += sh1[tid + st]; sh2[tid] += sh2[tid + st]; }
        __syncthreads();
    }
    float mean = sh1[0] / (float)count;
    float var  = sh2[0] / (float)count - mean * mean;
    float rstd = rsqrtf(var + 1e-5f);

    for (int i = tid; i < count; i += 256) {
        int c  = c0 + i / HWc;
        int hw = i % HWc;
        size_t o = ((size_t)c * Bc + b) * HWc + hw;
        float xn = (X[o] - mean) * rstd;
        float v  = xn * gamma[c] + beta[c];
        Y[o] = (v >= 0.f) ? v : 0.01f * v;
    }
}

// ---------------------------------------------------------------------------
// Cosine-sim weights + double-exp softmax + weighted sum + split output.
// tt,tk: [512][NT]; stt,stk: [256][NT]; out: concat([B,256,H,W],[B,256,H,W])
// ---------------------------------------------------------------------------
__global__ void stsn_fuse_out(const float* __restrict__ tt, const float* __restrict__ tk,
                              const float* __restrict__ stt, const float* __restrict__ stk,
                              float* __restrict__ out)
{
    int n = blockIdx.x * blockDim.x + threadIdx.x;
    if (n >= NT) return;
    int b = n / HWc, hw = n % HWc;
    float dtt = 0.f, dtk = 0.f, nkk = 0.f;
    for (int c = 0; c < 256; ++c) {
        float a = stt[(size_t)c * NT + n];
        float k = stk[(size_t)c * NT + n];
        dtt += a * a; dtk += a * k; nkk += k * k;
    }
    float na = sqrtf(dtt), nb = sqrtf(nkk);
    float cos_tt = dtt / fmaxf(na * na, 1e-8f);
    float cos_tk = dtk / fmaxf(na * nb, 1e-8f);
    float ttw = __expf(cos_tt), tkw = __expf(cos_tk);
    float e0 = __expf(ttw), e1 = __expf(tkw);
    float inv = 1.f / (e0 + e1);
    float w0 = e0 * inv, w1 = e1 * inv;
    const size_t part = (size_t)Bc * 256 * HWc;
    for (int c = 0; c < 512; ++c) {
        float v = w0 * tt[(size_t)c * NT + n] + w1 * tk[(size_t)c * NT + n];
        size_t o = (size_t)(c >> 8) * part + (size_t)b * 256 * HWc + (size_t)(c & 255) * HWc + hw;
        out[o] = v;
    }
}

// ---------------------------------------------------------------------------
// Host orchestration
// ---------------------------------------------------------------------------
extern "C" void kernel_launch(void* const* d_in, const int* in_sizes, int n_in,
                              void* d_out, int out_size, void* d_ws, size_t ws_size,
                              hipStream_t stream)
{
    (void)in_sizes; (void)n_in; (void)out_size; (void)ws_size;
    const float* datas = (const float*)d_in[0];
    const float* offw[4]  = { (const float*)d_in[1],  (const float*)d_in[6],
                              (const float*)d_in[11], (const float*)d_in[16] };
    const float* offb[4]  = { (const float*)d_in[2],  (const float*)d_in[7],
                              (const float*)d_in[12], (const float*)d_in[17] };
    const float* convw[4] = { (const float*)d_in[3],  (const float*)d_in[8],
                              (const float*)d_in[13], (const float*)d_in[18] };
    const float* gng[4]   = { (const float*)d_in[4],  (const float*)d_in[9],
                              (const float*)d_in[14], (const float*)d_in[19] };
    const float* gnb[4]   = { (const float*)d_in[5],  (const float*)d_in[10],
                              (const float*)d_in[15], (const float*)d_in[20] };
    const float* neck0_w = (const float*)d_in[21];
    const float* gn1_g   = (const float*)d_in[22];
    const float* gn1_b   = (const float*)d_in[23];
    const float* neck3_w = (const float*)d_in[24];
    const float* gn4_g   = (const float*)d_in[25];
    const float* gn4_b   = (const float*)d_in[26];
    const float* neck6_w = (const float*)d_in[27];
    float* out = (float*)d_out;

    char* wsp = (char*)d_ws;
    auto alloc = [&](size_t bytes) -> char* {
        char* p = wsp;
        wsp += (bytes + 255) & ~(size_t)255;
        return p;
    };
    float* f_buf    = (float*)alloc((size_t)1024 * NT * 4);
    float* actA     = (float*)alloc((size_t)1024 * NT * 4);
    float* actB     = (float*)alloc((size_t)1024 * NT * 4);
    float* gemm_out = (float*)alloc((size_t)1024 * NT * 4);
    float* offs     = (float*)alloc((size_t)18 * NT * 4);
    float* tt       = (float*)alloc((size_t)512 * NT * 4);
    float* tk       = (float*)alloc((size_t)512 * NT * 4);
    float* nk1      = (float*)alloc((size_t)512 * NT * 4);
    float* nk2      = (float*)alloc((size_t)256 * NT * 4);
    float* stt      = (float*)alloc((size_t)256 * NT * 4);
    float* stk      = (float*)alloc((size_t)256 * NT * 4);
    f16*   wf16     = (f16*)  alloc((size_t)1024 * 9216 * 2);
    f16*   Bmat     = (f16*)  alloc((size_t)NT * 9216 * 2);

    auto blocks = [](long total, int bs) { return (int)((total + bs - 1) / bs); };

    auto gemm = [&](const f16* A, const f16* Bm, float* C, int M, int K) {
        dim3 grid(NT / BN, M / BM);
        stsn_wmma_gemm<<<grid, 256, 0, stream>>>(A, Bm, C, M, NT, K);
    };

    auto run_agg = [&](int swap, float* aggdst) {
        stsn_build_concat<<<blocks((long)1024 * NT, 256), 256, 0, stream>>>(datas, f_buf, swap);
        const float* in = f_buf;
        float* acts[2] = { actA, actB };
        for (int L = 0; L < 4; ++L) {
            // offset conv on current 1024-ch activation
            stsn_conv3x3_direct<<<blocks((long)18 * NT, 256), 256, 0, stream>>>(
                in, offw[L], offb[L], offs, 1024, 18);
            // deform sampling: layers 0-2 sample `in`; layer 3 samples support = f[0:512]
            const float* ssrc = (L == 3) ? f_buf : in;
            int Cs = (L == 3) ? 512 : 1024;
            stsn_bilinear_im2col<<<blocks((long)9 * NT, 256), 256, 0, stream>>>(
                ssrc, offs, Bmat, Cs);
            int M = (L == 3) ? 512 : 1024;
            int K = Cs * 9;
            stsn_to_f16<<<blocks((long)M * K, 256), 256, 0, stream>>>(convw[L], wf16, (long)M * K);
            gemm(wf16, Bmat, gemm_out, M, K);
            float* dst = (L == 3) ? aggdst : acts[L & 1];
            stsn_gn_leaky<<<Bc * 32, 256, 0, stream>>>(gemm_out, dst, gng[L], gnb[L], M, M / 32);
            in = dst;
        }
    };

    auto run_neck = [&](const float* x, float* dst) {
        // neck0: 1x1 512->512
        stsn_to_f16_T<<<blocks((long)512 * NT, 256), 256, 0, stream>>>(x, Bmat, 512);
        stsn_to_f16<<<blocks(512L * 512, 256), 256, 0, stream>>>(neck0_w, wf16, 512L * 512);
        gemm(wf16, Bmat, gemm_out, 512, 512);
        stsn_gn_leaky<<<Bc * 32, 256, 0, stream>>>(gemm_out, nk1, gn1_g, gn1_b, 512, 16);
        // neck3: 3x3 512->256 pad 1
        stsn_im2col_regular<<<blocks((long)9 * NT, 256), 256, 0, stream>>>(nk1, Bmat, 512);
        stsn_to_f16<<<blocks(256L * 4608, 256), 256, 0, stream>>>(neck3_w, wf16, 256L * 4608);
        gemm(wf16, Bmat, gemm_out, 256, 4608);
        stsn_gn_leaky<<<Bc * 32, 256, 0, stream>>>(gemm_out, nk2, gn4_g, gn4_b, 256, 8);
        // neck6: 1x1 256->256
        stsn_to_f16_T<<<blocks((long)256 * NT, 256), 256, 0, stream>>>(nk2, Bmat, 256);
        stsn_to_f16<<<blocks(256L * 256, 256), 256, 0, stream>>>(neck6_w, wf16, 256L * 256);
        gemm(wf16, Bmat, dst, 256, 256);
    };

    run_agg(0, tt);
    run_neck(tt, stt);
    run_agg(1, tk);
    run_neck(tk, stk);
    stsn_fuse_out<<<blocks(NT, 256), 256, 0, stream>>>(tt, tk, stt, stk, out);
}
